// Attention_Vertical_4595615007193
// MI455X (gfx1250) — matrix operand order repaired and checked by reading, not compiled, not run on hardware
//
#include <hip/hip_runtime.h>
#include <hip/hip_bf16.h>

// ---------------------------------------------------------------------------
// Attention (B=8, N=1024, C=768, H=12, D=64) for gfx1250 (CDNA5, wave32).
// - All GEMMs: v_wmma_f32_16x16x32_f16 (f16 operands, fp32 accumulate).
// - Scores+softmax fused (16x1024 row-slab in LDS): raw scores never hit HBM.
// - A*V kernel stages the 402MB attn stream through LDS with the Tensor Data
//   Mover (tensor_load_to_lds, double-buffered, s_wait_tensorcnt handshake).
// - attn / out are write-once streams > L2 -> non-temporal stores.
// ---------------------------------------------------------------------------

typedef _Float16 half_t;
typedef __attribute__((ext_vector_type(16))) _Float16 v16h;
typedef __attribute__((ext_vector_type(8)))  _Float16 v8h;
typedef __attribute__((ext_vector_type(8)))  float    v8f;
typedef __attribute__((ext_vector_type(4)))  unsigned int v4u;
typedef __attribute__((ext_vector_type(8)))  int v8i;
typedef __attribute__((ext_vector_type(4)))  int v4i;

#define BATCH 8
#define SEQ   1024
#define CDIM  768
#define HEADS 12
#define HDIM  64
#define MROWS (BATCH * SEQ)          // 8192
#define SCALE 0.125f                 // 1/sqrt(64)

// ---- WMMA wrapper ----------------------------------------------------------
__device__ __forceinline__ v8f wmma_f16(v16h a, v16h b, v8f c) {
    return __builtin_amdgcn_wmma_f32_16x16x32_f16(
        /*neg_a=*/false, a, /*neg_b=*/false, b,
        /*c_mod=*/(short)0, c, /*reuse_a=*/false, /*reuse_b=*/false);
}

// ---- fragment loaders ------------------------------------------------------
// A (16x32, f16, row-major, leading dim ld), p points at tile origin.
// lane<16: row=lane,   K-chunks [0..7] and [16..23]
// lane>=16: row=lane-16, K-chunks [8..15] and [24..31]
__device__ __forceinline__ v16h load_a_f16(const half_t* p, int ld, int lane) {
    int row  = lane & 15;
    int koff = (lane >> 4) << 3;          // 0 or 8
    const half_t* q = p + row * ld + koff;
    v8h lo = *(const v8h*)(q);
    v8h hi = *(const v8h*)(q + 16);
    v16h a;
#pragma unroll
    for (int i = 0; i < 8; ++i) { a[i] = lo[i]; a[i + 8] = hi[i]; }
    return a;
}

// A (16x32) sourced from an fp32 tile staged in LDS (row-major, ld floats),
// converted to f16 in-register.  Emits ds_load_b128 pairs.
__device__ __forceinline__ v16h load_a_lds_f32(const float* p, int ld, int lane) {
    int row  = lane & 15;
    int koff = (lane >> 4) << 3;
    const float* q = p + row * ld + koff;
    v8f lo = *(const v8f*)(q);
    v8f hi = *(const v8f*)(q + 16);
    v16h a;
#pragma unroll
    for (int i = 0; i < 8; ++i) {
        a[i]     = (_Float16)lo[i];
        a[i + 8] = (_Float16)hi[i];
    }
    return a;
}

// B (32x16) for NT GEMM: B[k][n] = Wrow[n][k]; W row-major with leading dim ld.
// lane<16: col=lane, K=[0..7],[16..23] ; lane>=16: col=lane-16, K=[8..15],[24..31].
__device__ __forceinline__ v16h load_b_f16(const half_t* p, int ld, int lane) {
    int col  = lane & 15;
    int koff = (lane >> 4) << 3;          // 0 or 8
    const half_t* q = p + col * ld + koff;
    v8h lo = *(const v8h*)(q);
    v8h hi = *(const v8h*)(q + 16);
    v16h b;
#pragma unroll
    for (int i = 0; i < 8; ++i) { b[i] = lo[i]; b[i + 8] = hi[i]; }
    return b;
}

// ---- Tensor Data Mover: 2D fp32 tile, global -> LDS ------------------------
// D# packing per CDNA5 ISA sec.8:
//   group0: [1:0]=count=1 | lds_addr[63:32] | global_addr[120:64] | type=2
//   group1: data_size=4B, tensor dims, tile dims, dim0 stride
// This toolchain's 6-arg builtin form:
//   (uint32x4 g0, int32x8 g1, int32x4 g2, int32x4 g3, int32x8 gx, i32 cpol).
// Groups 2/3/x are unused for a 2D tile (tile_dim2 = 0).
__device__ __forceinline__ void tdm_load_2d_f32(
    unsigned lds_off, const float* gptr,
    unsigned tile_cols, unsigned tile_rows,
    unsigned tensor_cols, unsigned tensor_rows, unsigned row_stride) {
    unsigned long long ga = (unsigned long long)(size_t)gptr;
    v4u g0;
    g0[0] = 1u;                                         // count=1, user mode
    g0[1] = lds_off;                                    // lds_addr (bytes)
    g0[2] = (unsigned)(ga & 0xffffffffu);               // global_addr[31:0]
    g0[3] = (unsigned)((ga >> 32) & 0x1ffffffu)         // global_addr[56:32]
          | (2u << 30);                                 // type = 2 ("image")
    v8i g1;
    g1[0] = (int)(2u << 16);                            // data_size = 4 bytes
    g1[1] = (int)((tensor_cols & 0xffffu) << 16);       // tensor_dim0[15:0]
    g1[2] = (int)(((tensor_cols >> 16) & 0xffffu)       // tensor_dim0[31:16]
          |       ((tensor_rows & 0xffffu) << 16));     // tensor_dim1[15:0]
    g1[3] = (int)(((tensor_rows >> 16) & 0xffffu)       // tensor_dim1[31:16]
          |       ((tile_cols & 0xffffu) << 16));       // tile_dim0
    g1[4] = (int)(tile_rows & 0xffffu);                 // tile_dim1 (tile_dim2=0)
    g1[5] = (int)row_stride;                            // tensor_dim0_stride[31:0]
    g1[6] = 0;
    g1[7] = 0;
    v4i gz4 = {0, 0, 0, 0};                             // groups 2/3 unused (2D)
    v8i gz8 = {0, 0, 0, 0, 0, 0, 0, 0};                 // extra group unused
    __builtin_amdgcn_tensor_load_to_lds(g0, g1, gz4, gz4, gz8, 0);
}

// ---- kernel 1: fp32 -> fp16 convert ---------------------------------------
__global__ void cvt_f32_to_f16(const float* __restrict__ in,
                               half_t* __restrict__ out, int n) {
    int stride = gridDim.x * blockDim.x;
    for (int i = blockIdx.x * blockDim.x + threadIdx.x; i < n; i += stride)
        out[i] = (half_t)in[i];
}

// ---- kernel 2: fused QKV projection (NT GEMM, M=8192, N=768, K=768) -------
// blockIdx.z selects Wq/Wk/Wv. V output is stored transposed per head:
// Vt[b][h][d][tok]  (so the later A*V GEMM is the same NT fragment pattern).
__global__ void __launch_bounds__(256)
gemm_qkv(const half_t* __restrict__ xh,
         const half_t* __restrict__ Wqh, const half_t* __restrict__ Wkh,
         const half_t* __restrict__ Wvh,
         half_t* __restrict__ Qh, half_t* __restrict__ Kh,
         half_t* __restrict__ Vt) {
    const int z = blockIdx.z;
    const half_t* W = (z == 0) ? Wqh : (z == 1) ? Wkh : Wvh;

    const int tid  = threadIdx.x;
    const int lane = tid & 31;
    const int wave = tid >> 5;
    const int wm = wave & 3;              // 4 waves along M
    const int wn = wave >> 2;             // 2 waves along N
    const int m0 = blockIdx.x * 128 + wm * 32;
    const int n0 = blockIdx.y * 64  + wn * 32;

    v8f acc[2][2] = {};
    for (int k0 = 0; k0 < CDIM; k0 += 32) {
        v16h a0 = load_a_f16(xh + (size_t)m0 * CDIM + k0, CDIM, lane);
        v16h a1 = load_a_f16(xh + (size_t)(m0 + 16) * CDIM + k0, CDIM, lane);
        v16h b0 = load_b_f16(W + (size_t)n0 * CDIM + k0, CDIM, lane);
        v16h b1 = load_b_f16(W + (size_t)(n0 + 16) * CDIM + k0, CDIM, lane);
        acc[0][0] = wmma_f16(a0, b0, acc[0][0]);
        acc[0][1] = wmma_f16(a0, b1, acc[0][1]);
        acc[1][0] = wmma_f16(a1, b0, acc[1][0]);
        acc[1][1] = wmma_f16(a1, b1, acc[1][1]);
    }

    const int rbase = (lane >> 4) << 3;   // 0 or 8
    const int cbase = lane & 15;
#pragma unroll
    for (int mi = 0; mi < 2; ++mi)
#pragma unroll
        for (int ni = 0; ni < 2; ++ni)
#pragma unroll
            for (int r = 0; r < 8; ++r) {
                int m = m0 + mi * 16 + rbase + r;     // global row (b*1024+tok)
                int n = n0 + ni * 16 + cbase;         // feature
                half_t val = (half_t)acc[mi][ni][r];
                if (z == 0) {
                    Qh[(size_t)m * CDIM + n] = val;
                } else if (z == 1) {
                    Kh[(size_t)m * CDIM + n] = val;
                } else {
                    int b = m >> 10, tok = m & 1023;
                    int h = n >> 6,  d   = n & 63;
                    Vt[(((size_t)(b * HEADS + h)) * HDIM + d) * SEQ + tok] = val;
                }
            }
}

// ---- kernel 3: fused scores + softmax -------------------------------------
// One block owns rows [row0, row0+16) of one (b,h): S = Q K^T * scale, then
// row softmax over the full 1024 columns staged in 64KB LDS.
__global__ void __launch_bounds__(256)
attn_softmax(const half_t* __restrict__ Qh, const half_t* __restrict__ Kh,
             float* __restrict__ attn) {
    __shared__ float S[16 * SEQ];                 // 64 KB

    const int bh = blockIdx.y;
    const int b = bh / HEADS, h = bh % HEADS;
    const int row0 = blockIdx.x * 16;
    const int tid  = threadIdx.x;
    const int lane = tid & 31;
    const int wave = tid >> 5;

    const half_t* Qp = Qh + ((size_t)(b * SEQ + row0)) * CDIM + h * HDIM;

    v8f acc[8] = {};                              // wave covers 128 columns
    for (int k0 = 0; k0 < HDIM; k0 += 32) {
        v16h a = load_a_f16(Qp + k0, CDIM, lane);
#pragma unroll
        for (int t = 0; t < 8; ++t) {
            int m0 = wave * 128 + t * 16;
            v16h bf = load_b_f16(
                Kh + ((size_t)(b * SEQ + m0)) * CDIM + h * HDIM + k0,
                CDIM, lane);
            acc[t] = wmma_f16(a, bf, acc[t]);
        }
    }

    const int rbase = (lane >> 4) << 3;
    const int cbase = lane & 15;
#pragma unroll
    for (int t = 0; t < 8; ++t)
#pragma unroll
        for (int r = 0; r < 8; ++r)
            S[(rbase + r) * SEQ + wave * 128 + t * 16 + cbase] =
                acc[t][r] * SCALE;

    __syncthreads();

    // softmax: each wave handles 2 of the 16 rows
    float* orow_base = attn + ((size_t)bh * SEQ + row0) * SEQ;
#pragma unroll
    for (int rr = 0; rr < 2; ++rr) {
        int row = wave * 2 + rr;
        float* srow = S + row * SEQ;

        float mx = -3.4e38f;
        for (int c = lane; c < SEQ; c += 32) mx = fmaxf(mx, srow[c]);
#pragma unroll
        for (int off = 16; off > 0; off >>= 1)
            mx = fmaxf(mx, __shfl_xor(mx, off, 32));

        float sum = 0.f;
        for (int c = lane; c < SEQ; c += 32) {
            float e = __expf(srow[c] - mx);
            srow[c] = e;
            sum += e;
        }
#pragma unroll
        for (int off = 16; off > 0; off >>= 1)
            sum += __shfl_xor(sum, off, 32);

        float inv = 1.0f / sum;
        float* orow = orow_base + (size_t)row * SEQ;
        // attn is a 402MB write-once stream (> L2): non-temporal.
        for (int c = lane; c < SEQ; c += 32)
            __builtin_nontemporal_store(srow[c] * inv, &orow[c]);
    }
}

// ---- kernel 4: O = attn @ V (NT with Vt), TDM-staged A --------------------
// Per (b,h): attn[1024x1024] fp32 times Vt[64x1024] f16.
// The block's 128x32 attn tile for each k-step is DMA'd into LDS by the
// Tensor Data Mover (double-buffered); lanes read fragments via ds_load and
// convert to f16 in-register.  Wave 0 drives the DMA + tensorcnt handshake.
__global__ void __launch_bounds__(256)
av_gemm(const float* __restrict__ attn, const half_t* __restrict__ Vt,
        half_t* __restrict__ Oh) {
    __shared__ float S[2][128 * 32];              // 2 x 16KB tile buffers

    const int bh = blockIdx.y;
    const int b = bh / HEADS, h = bh % HEADS;
    const int tid  = threadIdx.x;
    const int lane = tid & 31;
    const int wave = tid >> 5;
    const int wm = wave & 3;
    const int wn = wave >> 2;
    const int tblk = blockIdx.x * 128;            // block's token-row base
    const int t0 = tblk + wm * 32;                // wave's token rows
    const int d0 = wn * 32;                       // head-dim cols (0 or 32)

    const float*  Ap = attn + (size_t)bh * SEQ * SEQ;
    const half_t* Bp = Vt   + (size_t)bh * HDIM * SEQ;

    const unsigned lds0 = (unsigned)(size_t)(void*)&S[0][0];
    const unsigned lds1 = (unsigned)(size_t)(void*)&S[1][0];

    // prologue: DMA tile for k0 = 0 into buffer 0
    if (wave == 0)
        tdm_load_2d_f32(lds0, Ap + (size_t)tblk * SEQ,
                        /*tile_cols=*/32, /*tile_rows=*/128,
                        /*tensor_cols=*/SEQ, /*tensor_rows=*/SEQ,
                        /*row_stride=*/SEQ);

    v8f acc[2][2] = {};
    int pb = 0;
    for (int k0 = 0; k0 < SEQ; k0 += 32, pb ^= 1) {
        if (wave == 0)
            __builtin_amdgcn_s_wait_tensorcnt(0);   // tile[pb] landed
        __syncthreads();                            // release all waves
        if (wave == 0 && k0 + 32 < SEQ)             // overlap next DMA
            tdm_load_2d_f32(pb ? lds0 : lds1,
                            Ap + (size_t)tblk * SEQ + (k0 + 32),
                            32, 128, SEQ, SEQ, SEQ);

        const float* At = &S[pb][0];                // 128 x 32, row-major
        v16h a0 = load_a_lds_f32(At + (wm * 32) * 32, 32, lane);
        v16h a1 = load_a_lds_f32(At + (wm * 32 + 16) * 32, 32, lane);
        v16h b0 = load_b_f16(Bp + (size_t)d0 * SEQ + k0, SEQ, lane);
        v16h b1 = load_b_f16(Bp + (size_t)(d0 + 16) * SEQ + k0, SEQ, lane);
        acc[0][0] = wmma_f16(a0, b0, acc[0][0]);
        acc[0][1] = wmma_f16(a0, b1, acc[0][1]);
        acc[1][0] = wmma_f16(a1, b0, acc[1][0]);
        acc[1][1] = wmma_f16(a1, b1, acc[1][1]);
        // next iteration's top barrier guards buffer reuse (double buffer)
    }

    const int rbase = (lane >> 4) << 3;
    const int cbase = lane & 15;
#pragma unroll
    for (int mi = 0; mi < 2; ++mi)
#pragma unroll
        for (int ni = 0; ni < 2; ++ni)
#pragma unroll
            for (int r = 0; r < 8; ++r) {
                int tok = t0 + mi * 16 + rbase + r;
                int d   = d0 + ni * 16 + cbase;
                Oh[((size_t)(b * SEQ + tok)) * CDIM + h * HDIM + d] =
                    (half_t)acc[mi][ni][r];
            }
}

// ---- kernel 5: out = Oh @ Wp^T + bp (fp32 output) -------------------------
__global__ void __launch_bounds__(256)
proj_gemm(const half_t* __restrict__ Oh, const half_t* __restrict__ Wph,
          const float* __restrict__ bp, float* __restrict__ out) {
    const int tid  = threadIdx.x;
    const int lane = tid & 31;
    const int wave = tid >> 5;
    const int wm = wave & 3;
    const int wn = wave >> 2;
    const int m0 = blockIdx.x * 128 + wm * 32;
    const int n0 = blockIdx.y * 64  + wn * 32;

    v8f acc[2][2] = {};
    for (int k0 = 0; k0 < CDIM; k0 += 32) {
        v16h a0 = load_a_f16(Oh + (size_t)m0 * CDIM + k0, CDIM, lane);
        v16h a1 = load_a_f16(Oh + (size_t)(m0 + 16) * CDIM + k0, CDIM, lane);
        v16h b0 = load_b_f16(Wph + (size_t)n0 * CDIM + k0, CDIM, lane);
        v16h b1 = load_b_f16(Wph + (size_t)(n0 + 16) * CDIM + k0, CDIM, lane);
        acc[0][0] = wmma_f16(a0, b0, acc[0][0]);
        acc[0][1] = wmma_f16(a0, b1, acc[0][1]);
        acc[1][0] = wmma_f16(a1, b0, acc[1][0]);
        acc[1][1] = wmma_f16(a1, b1, acc[1][1]);
    }

    const int rbase = (lane >> 4) << 3;
    const int cbase = lane & 15;
#pragma unroll
    for (int mi = 0; mi < 2; ++mi)
#pragma unroll
        for (int ni = 0; ni < 2; ++ni)
#pragma unroll
            for (int r = 0; r < 8; ++r) {
                int m = m0 + mi * 16 + rbase + r;
                int n = n0 + ni * 16 + cbase;
                __builtin_nontemporal_store(acc[mi][ni][r] + bp[n],
                                            &out[(size_t)m * CDIM + n]);
            }
}

// ---------------------------------------------------------------------------
extern "C" void kernel_launch(void* const* d_in, const int* in_sizes, int n_in,
                              void* d_out, int out_size, void* d_ws,
                              size_t ws_size, hipStream_t stream) {
    const float* x  = (const float*)d_in[0];
    const float* Wq = (const float*)d_in[1];
    const float* Wk = (const float*)d_in[2];
    const float* Wv = (const float*)d_in[3];
    const float* Wp = (const float*)d_in[4];
    const float* bp = (const float*)d_in[5];

    const size_t XN = (size_t)MROWS * CDIM;   // 6,291,456
    const size_t WN = (size_t)CDIM * CDIM;    // 589,824

    half_t* xh  = (half_t*)d_ws;
    half_t* Wqh = xh  + XN;
    half_t* Wkh = Wqh + WN;
    half_t* Wvh = Wkh + WN;
    half_t* Wph = Wvh + WN;
    half_t* Qh  = Wph + WN;
    half_t* Kh  = Qh  + XN;
    half_t* Vt  = Kh  + XN;                   // [B,H,64,1024]
    half_t* Oh  = Vt  + XN;

    float* out  = (float*)d_out;              // [B,N,C]
    float* attn = out + XN;                   // [B,H,N,N]

    // 1) fp32 -> fp16 conversions
    cvt_f32_to_f16<<<1024, 256, 0, stream>>>(x,  xh,  (int)XN);
    cvt_f32_to_f16<<<288,  256, 0, stream>>>(Wq, Wqh, (int)WN);
    cvt_f32_to_f16<<<288,  256, 0, stream>>>(Wk, Wkh, (int)WN);
    cvt_f32_to_f16<<<288,  256, 0, stream>>>(Wv, Wvh, (int)WN);
    cvt_f32_to_f16<<<288,  256, 0, stream>>>(Wp, Wph, (int)WN);

    // 2) Q,K,V projections (V stored transposed per head)
    gemm_qkv<<<dim3(MROWS / 128, CDIM / 64, 3), 256, 0, stream>>>(
        xh, Wqh, Wkh, Wvh, Qh, Kh, Vt);

    // 3) fused scores + softmax -> attn (fp32, part of d_out)
    attn_softmax<<<dim3(SEQ / 16, BATCH * HEADS), 256, 0, stream>>>(
        Qh, Kh, attn);

    // 4) O = attn @ V  (attn tiles streamed via Tensor Data Mover)
    av_gemm<<<dim3(SEQ / 128, BATCH * HEADS), 256, 0, stream>>>(attn, Vt, Oh);

    // 5) out = O @ Wp^T + bp
    proj_gemm<<<dim3(MROWS / 128, CDIM / 64), 256, 0, stream>>>(
        Oh, Wph, bp, out);
}
